// TriUpdate_49727131353914
// MI455X (gfx1250) — matrix operand order repaired and checked by reading, not compiled, not run on hardware
//
#include <hip/hip_runtime.h>
#include <hip/hip_bf16.h>

typedef __bf16 bf16_t;
typedef __attribute__((ext_vector_type(16))) __bf16 v16bf;
typedef __attribute__((ext_vector_type(8)))  float  v8f;

#define LSEQ 256
#define DMODEL 128
#define DMHALF 64
#define NHEAD 4
#define DHEAD 32
#define NTOK (LSEQ * LSEQ)   // 65536

// ---------------------------------------------------------------------------
// WMMA helpers (CDNA5 16x16x32 bf16, fp32 accumulate)
// ---------------------------------------------------------------------------
union U16V { uint4 u[2]; v16bf v; };

static __device__ __forceinline__ v8f splat8(float x) {
  v8f v = {x, x, x, x, x, x, x, x};
  return v;
}

// A-matrix 16x32 bf16 fragment. base points at (row0, k0) of a row-major
// [M,K] bf16 matrix with leading dimension lda (halfs).
// Lane l (l<16): M=l, K runs {k0..k0+7, k0+16..k0+23}; lanes 16-31: +8.
static __device__ __forceinline__ v16bf frag_a(const bf16_t* base, int lda, int lane) {
  int hl = lane & 15, hi = lane >> 4;
  const bf16_t* r = base + (long)hl * lda + hi * 8;
  U16V f;
  f.u[0] = *(const uint4*)(r);
  f.u[1] = *(const uint4*)(r + 16);
  return f.v;
}

// B-matrix 32x16 bf16 fragment from a K-major ("Bt") [N,K] layout.
// Lane l (l<16): N=l, K runs {k0..k0+7, k0+16..k0+23}; lanes 16-31: +8.
static __device__ __forceinline__ v16bf frag_b(const bf16_t* base, int ldb, int lane) {
  int hl = lane & 15, hi = lane >> 4;
  const bf16_t* r = base + (long)hl * ldb + hi * 8;
  U16V f;
  f.u[0] = *(const uint4*)(r);
  f.u[1] = *(const uint4*)(r + 16);
  return f.v;
}

static __device__ __forceinline__ v8f wmma_bf16(v16bf a, v16bf b, v8f c) {
  return __builtin_amdgcn_wmma_f32_16x16x32_bf16(false, a, false, b, (short)0, c,
                                                 false, false);
}

static __device__ __forceinline__ float sigf(float x) {
  return 1.0f / (1.0f + __expf(-x));
}

// ---------------------------------------------------------------------------
// Generic WMMA GEMM, register-blocked: each wave computes a 16x64 output tile
// (4 accumulators) so every A fragment load feeds 4 WMMAs.
// out[z, m, n] = epilogue( A[z][m,:K] . Bt[z][n,:K] + bias[n] )
// modes: 0 = store fp32, 1 = store sigmoid fp32, 2 = accumulate (+=) fp32,
//        3 = store bf16
// ---------------------------------------------------------------------------
__global__ __launch_bounds__(32) void gemm_wmma_kernel(
    const bf16_t* __restrict__ A, int lda, long aZ,
    const bf16_t* __restrict__ Bt, int ldb, long bZ,
    const float* __restrict__ bias,
    void* __restrict__ out, long out_rs, long out_cs, long oZ,
    int K, int mode) {
  int lane = threadIdx.x;
  int tm = blockIdx.x, tn = blockIdx.y, z = blockIdx.z;
  const bf16_t* a = A + (long)z * aZ + (long)tm * 16 * lda;
  const bf16_t* b = Bt + (long)z * bZ + (long)tn * 64 * ldb;
  int hl = lane & 15, hi = lane >> 4;

  v8f acc[4];
#pragma unroll
  for (int s = 0; s < 4; ++s)
    acc[s] = splat8(bias ? bias[tn * 64 + s * 16 + hl] : 0.0f);

  for (int k = 0; k < K; k += 32) {
    __builtin_prefetch(a + k + 128, 0, 0);   // global_prefetch_b8
    v16bf fa = frag_a(a + k, lda, lane);     // reused by 4 WMMAs
#pragma unroll
    for (int s = 0; s < 4; ++s) {
      v16bf fb = frag_b(b + (long)(s * 16) * ldb + k, ldb, lane);
      acc[s] = wmma_bf16(fa, fb, acc[s]);
    }
  }
  // C/D layout: VGPR r -> row = r + (lane>=16)*8, col = lane&15
#pragma unroll
  for (int s = 0; s < 4; ++s) {
    for (int r = 0; r < 8; ++r) {
      long row = (long)tm * 16 + r + hi * 8;
      long col = (long)tn * 64 + s * 16 + hl;
      long idx = (long)z * oZ + row * out_rs + col * out_cs;
      float v = acc[s][r];
      if (mode == 1) {
        ((float*)out)[idx] = sigf(v);
      } else if (mode == 2) {
        ((float*)out)[idx] += v;
      } else if (mode == 3) {
        ((bf16_t*)out)[idx] = (bf16_t)v;
      } else {
        ((float*)out)[idx] = v;
      }
    }
  }
}

// ---------------------------------------------------------------------------
// LayerNorm over last dim (Dd = 128 or 64), one wave per token, bf16 output.
// ---------------------------------------------------------------------------
__global__ __launch_bounds__(256) void ln_bf16_kernel(
    const float* __restrict__ x, const float* __restrict__ g,
    const float* __restrict__ b, bf16_t* __restrict__ out, int ntok, int Dd) {
  int wave = threadIdx.x >> 5;
  int lane = threadIdx.x & 31;
  int tok = blockIdx.x * 8 + wave;
  if (tok >= ntok) return;
  int vec = Dd >> 5;  // 4 for D=128, 2 for D=64
  float v[4];
  const float* xp = x + (long)tok * Dd + lane * vec;
  float s = 0.0f;
  for (int i = 0; i < vec; ++i) { v[i] = xp[i]; s += v[i]; }
  for (int m = 16; m >= 1; m >>= 1) s += __shfl_xor(s, m, 32);
  float mean = s / (float)Dd;
  float t = 0.0f;
  for (int i = 0; i < vec; ++i) { float d = v[i] - mean; t += d * d; }
  for (int m = 16; m >= 1; m >>= 1) t += __shfl_xor(t, m, 32);
  float rstd = rsqrtf(t / (float)Dd + 1e-5f);
  bf16_t* op = out + (long)tok * Dd + lane * vec;
  const float* gp = g + lane * vec;
  const float* bp = b + lane * vec;
  for (int i = 0; i < vec; ++i)
    op[i] = (bf16_t)((v[i] - mean) * rstd * gp[i] + bp[i]);
}

// ---------------------------------------------------------------------------
// Elementwise kernels
// ---------------------------------------------------------------------------
__global__ __launch_bounds__(256) void copy_f4_kernel(const float4* __restrict__ in,
                                                      float4* __restrict__ out, int n4) {
  int i = blockIdx.x * 256 + threadIdx.x;
  if (i < n4) out[i] = in[i];
}

// fp32 [K,N] weight -> bf16 [N,K] (K-major for B fragments)
__global__ __launch_bounds__(256) void wtrans_kernel(const float* __restrict__ src,
                                                     bf16_t* __restrict__ dst,
                                                     int K, int N) {
  int i = blockIdx.x * 256 + threadIdx.x;
  if (i >= K * N) return;
  int k = i / N, n = i - k * N;
  dst[(long)n * K + k] = (bf16_t)src[i];
}

// ab = sigmoid(p2) * p1 ; split into a (cols 0..63) / b (cols 64..127), bf16
__global__ __launch_bounds__(256) void combine_ab_kernel(
    const float* __restrict__ p1, const float* __restrict__ p2,
    bf16_t* __restrict__ abuf, bf16_t* __restrict__ bbuf) {
  int i = blockIdx.x * 256 + threadIdx.x;
  if (i >= NTOK * DMODEL) return;
  int tok = i >> 7, c = i & 127;
  float v = sigf(p2[i]) * p1[i];
  if (c < DMHALF) abuf[(long)tok * DMHALF + c] = (bf16_t)v;
  else            bbuf[(long)tok * DMHALF + (c - DMHALF)] = (bf16_t)v;
}

// Channel-major transpose for the triangle einsum.
// dir 0 (outgoing): out[d][i][k] = in[(i*L+k)*DM + d]
// dir 1 (incoming): out[d][i][k] = in[(k*L+i)*DM + d]
__global__ __launch_bounds__(256) void tri_transpose_kernel(
    const bf16_t* __restrict__ in, bf16_t* __restrict__ out, int dir) {
  int idx = blockIdx.x * 256 + threadIdx.x;
  if (idx >= DMHALF * LSEQ * LSEQ) return;
  int d = idx >> 16;
  int rem = idx & 65535;
  int i = rem >> 8, k = rem & 255;
  long tok = dir ? ((long)k * LSEQ + i) : ((long)i * LSEQ + k);
  out[idx] = in[tok * DMHALF + d];
}

// zcur += gate * t
__global__ __launch_bounds__(256) void gate_madd_kernel(
    float* __restrict__ zcur, const float* __restrict__ gate,
    const float* __restrict__ t, int n) {
  int i = blockIdx.x * 256 + threadIdx.x;
  if (i < n) zcur[i] += gate[i] * t[i];
}

// gated = bf16(gate * t)
__global__ __launch_bounds__(256) void gate_mul_bf16_kernel(
    const float* __restrict__ gate, const float* __restrict__ t,
    bf16_t* __restrict__ out, int n) {
  int i = blockIdx.x * 256 + threadIdx.x;
  if (i < n) out[i] = (bf16_t)(gate[i] * t[i]);
}

// pair bias: bp[tok][h] = zn[tok,:].wp[:,h]   (N=4, tiny -> VALU dots)
__global__ __launch_bounds__(256) void bp_kernel(const bf16_t* __restrict__ zn,
                                                 const float* __restrict__ wp,
                                                 float* __restrict__ bp) {
  int i = blockIdx.x * 256 + threadIdx.x;
  if (i >= NTOK * NHEAD) return;
  int tok = i >> 2, h = i & 3;
  const bf16_t* zr = zn + (long)tok * DMODEL;
  float acc = 0.0f;
  for (int k = 0; k < DMODEL; ++k) acc += (float)zr[k] * wp[k * NHEAD + h];
  bp[i] = acc;
}

// V transpose for attention: vt[((fx*H+h)*DH+d)*L + j] = qkv[tok(fx,j)][256+h*32+d]
__global__ __launch_bounds__(256) void vt_build_kernel(const bf16_t* __restrict__ qkv,
                                                       bf16_t* __restrict__ vt,
                                                       int fsTok, int ffTok) {
  int idx = blockIdx.x * 256 + threadIdx.x;
  if (idx >= LSEQ * NHEAD * DHEAD * LSEQ) return;
  int j = idx & 255;
  int d = (idx >> 8) & 31;
  int h = (idx >> 13) & 3;
  int fx = idx >> 15;
  long tok = (long)fx * fsTok + (long)j * ffTok;
  vt[idx] = qkv[tok * 384 + 256 + h * 32 + d];
}

// ---------------------------------------------------------------------------
// Triangle attention core: one wave per (query 16-tile, fixed index, head).
// Scores via WMMA -> LDS, per-row softmax, P (bf16, LDS) @ V via WMMA.
// tok(fx, free) = fx*fsTok + free*ffTok.
// ---------------------------------------------------------------------------
__global__ __launch_bounds__(32) void attn_core_kernel(
    const bf16_t* __restrict__ qkv, const float* __restrict__ bp,
    const bf16_t* __restrict__ vt, float* __restrict__ attnout,
    int fsTok, int ffTok) {
  __shared__ __align__(16) float  s[16 * LSEQ];
  __shared__ __align__(16) bf16_t p[16 * LSEQ];
  int lane = threadIdx.x;
  int it = blockIdx.x, fx = blockIdx.y, h = blockIdx.z;
  int hl = lane & 15, hi = lane >> 4;
  int qlda = ffTok * 384;

  // Q fragment: rows = 16 queries, K = DH = 32 (single fragment, reused 16x)
  const bf16_t* qb = qkv + ((long)fx * fsTok + (long)it * 16 * ffTok) * 384 + h * 32;
  v16bf fq = frag_a(qb, qlda, lane);

  const float rscale = 0.1767766952966369f;  // 1/sqrt(32)
  for (int jt = 0; jt < 16; ++jt) {
    const bf16_t* kb =
        qkv + ((long)fx * fsTok + (long)jt * 16 * ffTok) * 384 + 128 + h * 32;
    v16bf fk = frag_b(kb, qlda, lane);
    v8f c = wmma_bf16(fq, fk, splat8(0.0f));
    for (int r = 0; r < 8; ++r) {
      int m = r + hi * 8;
      int j = jt * 16 + hl;
      float bias = bp[((long)(it * 16 + m) * LSEQ + j) * NHEAD + h];
      s[m * LSEQ + j] = c[r] * rscale + bias;
    }
  }
  __syncthreads();

  if (lane < 16) {
    float mx = -1e30f;
    for (int j = 0; j < LSEQ; ++j) mx = fmaxf(mx, s[lane * LSEQ + j]);
    float sum = 0.0f;
    for (int j = 0; j < LSEQ; ++j) {
      float e = __expf(s[lane * LSEQ + j] - mx);
      s[lane * LSEQ + j] = e;
      sum += e;
    }
    float inv = 1.0f / sum;
    for (int j = 0; j < LSEQ; ++j)
      p[lane * LSEQ + j] = (bf16_t)(s[lane * LSEQ + j] * inv);
  }
  __syncthreads();

  // out[16 x 32] = P[16 x 256] @ V[256 x 32]; one ds P-fragment feeds 2 WMMAs.
  const bf16_t* vb_base = vt + (((long)fx * NHEAD + h) * DHEAD) * LSEQ;
  v8f acc[2];
  acc[0] = splat8(0.0f);
  acc[1] = splat8(0.0f);
  for (int kc = 0; kc < 8; ++kc) {
    v16bf fp = frag_a(p + kc * 32, LSEQ, lane);               // ds_load_b128
#pragma unroll
    for (int nt = 0; nt < 2; ++nt) {
      v16bf fv = frag_b(vb_base + (long)nt * 16 * LSEQ + kc * 32, LSEQ, lane);
      acc[nt] = wmma_bf16(fp, fv, acc[nt]);
    }
  }
#pragma unroll
  for (int nt = 0; nt < 2; ++nt) {
    for (int r = 0; r < 8; ++r) {
      int i = it * 16 + r + hi * 8;
      int d = nt * 16 + hl;
      long tok = (long)fx * fsTok + (long)i * ffTok;
      attnout[tok * DMODEL + h * 32 + d] = acc[nt][r];
    }
  }
}

// ---------------------------------------------------------------------------
// Fused pair transition: hidden = relu(zn@w1+b1) [16x512 in LDS, bf16],
// zcur += hidden@w2 + b2. One wave per 16-token tile.
// Stage 1 hoists the 4 A fragments (each feeds 32 WMMAs); stage 2 inverts the
// loop nest so each LDS hidden fragment feeds 8 WMMAs.
// ---------------------------------------------------------------------------
__global__ __launch_bounds__(32) void trans_fused_kernel(
    const bf16_t* __restrict__ zn, const bf16_t* __restrict__ w1t,
    const float* __restrict__ b1, const bf16_t* __restrict__ w2t,
    const float* __restrict__ b2, float* __restrict__ zcur) {
  __shared__ __align__(16) bf16_t hb[16 * 512];
  int lane = threadIdx.x;
  int mt = blockIdx.x;
  int hl = lane & 15, hi = lane >> 4;
  const bf16_t* a = zn + (long)mt * 16 * DMODEL;

  v16bf fa[4];
#pragma unroll
  for (int kc = 0; kc < 4; ++kc) fa[kc] = frag_a(a + kc * 32, DMODEL, lane);

  for (int nt = 0; nt < 32; ++nt) {
    v8f acc = splat8(b1[nt * 16 + hl]);
#pragma unroll
    for (int kc = 0; kc < 4; ++kc) {
      v16bf fb = frag_b(w1t + (long)(nt * 16) * DMODEL + kc * 32, DMODEL, lane);
      acc = wmma_bf16(fa[kc], fb, acc);
    }
    for (int r = 0; r < 8; ++r)
      hb[(r + hi * 8) * 512 + nt * 16 + hl] = (bf16_t)fmaxf(acc[r], 0.0f);
  }
  __syncthreads();

  v8f acc2[8];
#pragma unroll
  for (int nt = 0; nt < 8; ++nt) acc2[nt] = splat8(b2[nt * 16 + hl]);
  for (int kc = 0; kc < 16; ++kc) {
    v16bf fh = frag_a(hb + kc * 32, 512, lane);               // ds_load_b128
#pragma unroll
    for (int nt = 0; nt < 8; ++nt) {
      v16bf fb = frag_b(w2t + (long)(nt * 16) * 512 + kc * 32, 512, lane);
      acc2[nt] = wmma_bf16(fh, fb, acc2[nt]);
    }
  }
#pragma unroll
  for (int nt = 0; nt < 8; ++nt)
    for (int r = 0; r < 8; ++r)
      zcur[((long)mt * 16 + r + hi * 8) * DMODEL + nt * 16 + hl] += acc2[nt][r];
}

// ---------------------------------------------------------------------------
// Host orchestration
// ---------------------------------------------------------------------------
extern "C" void kernel_launch(void* const* d_in, const int* in_sizes, int n_in,
                              void* d_out, int out_size, void* d_ws, size_t ws_size,
                              hipStream_t stream) {
  (void)in_sizes; (void)n_in; (void)out_size; (void)ws_size;
  auto F = [&](int i) { return (const float*)d_in[i]; };

  // --- workspace layout (bump allocator, 256B aligned) ---
  char* ws = (char*)d_ws;
  size_t off = 0;
  auto alloc = [&](size_t bytes) -> void* {
    void* p = ws + off;
    off = (off + bytes + 255) & ~(size_t)255;
    return p;
  };
  float*  zcur = (float*)alloc((size_t)NTOK * DMODEL * 4);
  bf16_t* zn   = (bf16_t*)alloc((size_t)NTOK * DMODEL * 2);
  float*  gate = (float*)alloc((size_t)NTOK * DMODEL * 4);
  float*  t0   = (float*)alloc((size_t)NTOK * DMODEL * 4);
  float*  t1   = (float*)alloc((size_t)NTOK * DMODEL * 4);
  // transposed bf16 weights
  bf16_t* w_mo_w1 = (bf16_t*)alloc(128 * 128 * 2);
  bf16_t* w_mo_w2 = (bf16_t*)alloc(128 * 128 * 2);
  bf16_t* w_mo_wg = (bf16_t*)alloc(128 * 128 * 2);
  bf16_t* w_mo_wo = (bf16_t*)alloc(128 * 64 * 2);
  bf16_t* w_mi_w1 = (bf16_t*)alloc(128 * 128 * 2);
  bf16_t* w_mi_w2 = (bf16_t*)alloc(128 * 128 * 2);
  bf16_t* w_mi_wg = (bf16_t*)alloc(128 * 128 * 2);
  bf16_t* w_mi_wo = (bf16_t*)alloc(128 * 64 * 2);
  bf16_t* w_ra_qkv = (bf16_t*)alloc(384 * 128 * 2);
  bf16_t* w_ra_wg  = (bf16_t*)alloc(128 * 128 * 2);
  bf16_t* w_ra_wo  = (bf16_t*)alloc(128 * 128 * 2);
  bf16_t* w_ca_qkv = (bf16_t*)alloc(384 * 128 * 2);
  bf16_t* w_ca_wg  = (bf16_t*)alloc(128 * 128 * 2);
  bf16_t* w_ca_wo  = (bf16_t*)alloc(128 * 128 * 2);
  bf16_t* w_tr_w1  = (bf16_t*)alloc(512 * 128 * 2);
  bf16_t* w_tr_w2  = (bf16_t*)alloc(128 * 512 * 2);
  // phase-aliased region (mult staging vs attention staging)
  char* region = (char*)alloc(90u * 1024u * 1024u);
  // mult phase views
  bf16_t* abuf = (bf16_t*)(region);
  bf16_t* bbuf = (bf16_t*)(region + 8388608);
  bf16_t* at   = (bf16_t*)(region + 2 * 8388608);
  bf16_t* bt   = (bf16_t*)(region + 3 * 8388608);
  float*  prod = (float*)(region + 4 * 8388608);
  bf16_t* pn   = (bf16_t*)(region + 4 * 8388608 + 16777216);
  // attention phase views
  bf16_t* qkv   = (bf16_t*)(region);
  float*  bpb   = (float*)(region + 50331648);
  bf16_t* vt    = (bf16_t*)(region + 50331648 + 1048576);
  bf16_t* gated = (bf16_t*)(region + 50331648 + 1048576 + 16777216);

  const int EW = NTOK * DMODEL;                 // 8388608
  dim3 B256(256);
  auto blks = [](int n) { return dim3((n + 255) / 256); };

  // z -> zcur
  copy_f4_kernel<<<blks(EW / 4), B256, 0, stream>>>((const float4*)F(0),
                                                    (float4*)zcur, EW / 4);
  // weight transposes
  wtrans_kernel<<<blks(128 * 128), B256, 0, stream>>>(F(3),  w_mo_w1, 128, 128);
  wtrans_kernel<<<blks(128 * 128), B256, 0, stream>>>(F(5),  w_mo_w2, 128, 128);
  wtrans_kernel<<<blks(128 * 128), B256, 0, stream>>>(F(7),  w_mo_wg, 128, 128);
  wtrans_kernel<<<blks(64 * 128),  B256, 0, stream>>>(F(11), w_mo_wo, 64, 128);
  wtrans_kernel<<<blks(128 * 128), B256, 0, stream>>>(F(15), w_mi_w1, 128, 128);
  wtrans_kernel<<<blks(128 * 128), B256, 0, stream>>>(F(17), w_mi_w2, 128, 128);
  wtrans_kernel<<<blks(128 * 128), B256, 0, stream>>>(F(19), w_mi_wg, 128, 128);
  wtrans_kernel<<<blks(64 * 128),  B256, 0, stream>>>(F(23), w_mi_wo, 64, 128);
  wtrans_kernel<<<blks(128 * 384), B256, 0, stream>>>(F(27), w_ra_qkv, 128, 384);
  wtrans_kernel<<<blks(128 * 128), B256, 0, stream>>>(F(29), w_ra_wg, 128, 128);
  wtrans_kernel<<<blks(128 * 128), B256, 0, stream>>>(F(31), w_ra_wo, 128, 128);
  wtrans_kernel<<<blks(128 * 384), B256, 0, stream>>>(F(35), w_ca_qkv, 128, 384);
  wtrans_kernel<<<blks(128 * 128), B256, 0, stream>>>(F(37), w_ca_wg, 128, 128);
  wtrans_kernel<<<blks(128 * 128), B256, 0, stream>>>(F(39), w_ca_wo, 128, 128);
  wtrans_kernel<<<blks(128 * 512), B256, 0, stream>>>(F(43), w_tr_w1, 128, 512);
  wtrans_kernel<<<blks(512 * 128), B256, 0, stream>>>(F(45), w_tr_w2, 512, 128);

  dim3 W32(32);
  dim3 gTok128(NTOK / 16, 128 / 64, 1);   // 16x64 tile per wave
  dim3 gTok384(NTOK / 16, 384 / 64, 1);
  dim3 gTri(LSEQ / 16, LSEQ / 64, DMHALF);

  // ---- triangular multiplicative update ----
  auto tri_mult = [&](int pb, bf16_t* w1t, bf16_t* w2t, bf16_t* wgt, bf16_t* wot,
                      int dir) {
    ln_bf16_kernel<<<blks(NTOK * 32) , B256, 0, stream>>>(zcur, F(pb + 0), F(pb + 1),
                                                          zn, NTOK, DMODEL);
    gemm_wmma_kernel<<<gTok128, W32, 0, stream>>>(zn, DMODEL, 0, w1t, DMODEL, 0,
        F(pb + 3), t0, DMODEL, 1, 0, DMODEL, 0);
    gemm_wmma_kernel<<<gTok128, W32, 0, stream>>>(zn, DMODEL, 0, w2t, DMODEL, 0,
        F(pb + 5), t1, DMODEL, 1, 0, DMODEL, 0);
    gemm_wmma_kernel<<<gTok128, W32, 0, stream>>>(zn, DMODEL, 0, wgt, DMODEL, 0,
        F(pb + 7), gate, DMODEL, 1, 0, DMODEL, 1);
    combine_ab_kernel<<<blks(EW), B256, 0, stream>>>(t0, t1, abuf, bbuf);
    tri_transpose_kernel<<<blks(DMHALF * NTOK), B256, 0, stream>>>(abuf, at, dir);
    tri_transpose_kernel<<<blks(DMHALF * NTOK), B256, 0, stream>>>(bbuf, bt, dir);
    // prod[i,j,d] = sum_k at[d][i][k]*bt[d][j][k]   (64-batch WMMA GEMM)
    gemm_wmma_kernel<<<gTri, W32, 0, stream>>>(at, LSEQ, (long)LSEQ * LSEQ,
        bt, LSEQ, (long)LSEQ * LSEQ, nullptr,
        prod, (long)LSEQ * DMHALF, DMHALF, 1, LSEQ, 0);
    ln_bf16_kernel<<<blks(NTOK * 32), B256, 0, stream>>>(prod, F(pb + 8), F(pb + 9),
                                                         pn, NTOK, DMHALF);
    gemm_wmma_kernel<<<gTok128, W32, 0, stream>>>(pn, DMHALF, 0, wot, DMHALF, 0,
        F(pb + 11), t0, DMODEL, 1, 0, DMHALF, 0);
    gate_madd_kernel<<<blks(EW), B256, 0, stream>>>(zcur, gate, t0, EW);
  };
  tri_mult(1, w_mo_w1, w_mo_w2, w_mo_wg, w_mo_wo, 0);   // outgoing
  tri_mult(13, w_mi_w1, w_mi_w2, w_mi_wg, w_mi_wo, 1);  // incoming

  // ---- triangle attention ----
  auto tri_attn = [&](int pb, bf16_t* qkvt, bf16_t* wgt, bf16_t* wot,
                      int fsTok, int ffTok) {
    ln_bf16_kernel<<<blks(NTOK * 32), B256, 0, stream>>>(zcur, F(pb + 0), F(pb + 1),
                                                         zn, NTOK, DMODEL);
    gemm_wmma_kernel<<<gTok384, W32, 0, stream>>>(zn, DMODEL, 0, qkvt, DMODEL, 0,
        nullptr, qkv, 384, 1, 0, DMODEL, 3);
    bp_kernel<<<blks(NTOK * NHEAD), B256, 0, stream>>>(zn, F(pb + 3), bpb);
    gemm_wmma_kernel<<<gTok128, W32, 0, stream>>>(zn, DMODEL, 0, wgt, DMODEL, 0,
        F(pb + 5), gate, DMODEL, 1, 0, DMODEL, 1);
    vt_build_kernel<<<blks(LSEQ * NHEAD * DHEAD * LSEQ), B256, 0, stream>>>(
        qkv, vt, fsTok, ffTok);
    dim3 gAttn(LSEQ / 16, LSEQ, NHEAD);
    attn_core_kernel<<<gAttn, W32, 0, stream>>>(qkv, bpb, vt, t0, fsTok, ffTok);
    gate_mul_bf16_kernel<<<blks(EW), B256, 0, stream>>>(gate, t0, gated, EW);
    gemm_wmma_kernel<<<gTok128, W32, 0, stream>>>(gated, DMODEL, 0, wot, DMODEL, 0,
        F(pb + 7), zcur, DMODEL, 1, 0, DMODEL, 2);
  };
  tri_attn(25, w_ra_qkv, w_ra_wg, w_ra_wo, 256, 1);   // row-wise
  tri_attn(33, w_ca_qkv, w_ca_wg, w_ca_wo, 1, 256);   // column-wise

  // ---- pair transition (fused MLP) ----
  ln_bf16_kernel<<<blks(NTOK * 32), B256, 0, stream>>>(zcur, F(41), F(42), zn,
                                                       NTOK, DMODEL);
  trans_fused_kernel<<<dim3(NTOK / 16), W32, 0, stream>>>(zn, w_tr_w1, F(44),
                                                          w_tr_w2, F(46), zcur);

  // zcur -> d_out
  copy_f4_kernel<<<blks(EW / 4), B256, 0, stream>>>((const float4*)zcur,
                                                    (float4*)d_out, EW / 4);
}
